// Head_85839216378204
// MI455X (gfx1250) — matrix operand order repaired and checked by reading, not compiled, not run on hardware
//
#include <hip/hip_runtime.h>
#include <hip/hip_bf16.h>
#include <math.h>

// ---------------------------------------------------------------------------
// Fused attention head for MI455X (gfx1250, wave32, WMMA).
//   Kernel 0: transpose/convert Wk,Wq,Wv (f32 1024x64) -> Wt3 f16 [3][64][1024]
//   Kernel 1: fused QKV projection: x is streamed ONCE; W chunks staged into
//             LDS with double-buffered global_load_async_to_lds_b128.
//   Kernel 2: flash attention (online softmax, WMMA f16 for QK^T and PV).
// Workspace: Q,K (f16 row-major) + V^T (f16) = 6 MB, + Wt3 = 384 KB.
// ---------------------------------------------------------------------------

#define CTX    2048
#define DMODEL 1024
#define HEADD  64
#define NB     8
#define CCH    128            // contraction chunk staged in LDS per buffer

typedef __attribute__((ext_vector_type(16))) _Float16 v16h;
typedef __attribute__((ext_vector_type(8)))  _Float16 v8h;
typedef __attribute__((ext_vector_type(8)))  float    v8f;
typedef __attribute__((ext_vector_type(4)))  float    f32x4;

static __device__ __forceinline__ v8f vzero8() {
  v8f z;
#pragma unroll
  for (int i = 0; i < 8; ++i) z[i] = 0.0f;
  return z;
}

static __device__ __forceinline__ v8f wmma_f16(v16h a, v16h b, v8f c) {
  // D = A(16x32 f16) * B(32x16 f16) + C(16x16 f32)
  return __builtin_amdgcn_wmma_f32_16x16x32_f16(false, a, false, b,
                                                (short)0, c, false, false);
}

// reduce across the 16 lanes of a half-wave (xor masks < 16 stay in-half)
static __device__ __forceinline__ float rowmax16(float v) {
#pragma unroll
  for (int m = 1; m < 16; m <<= 1) v = fmaxf(v, __shfl_xor(v, m, 32));
  return v;
}
static __device__ __forceinline__ float rowsum16(float v) {
#pragma unroll
  for (int m = 1; m < 16; m <<= 1) v += __shfl_xor(v, m, 32);
  return v;
}

// ---------------------------------------------------------------------------
// Kernel 0: one-time W transpose+convert into workspace:
//   Wt3[p][h][c] = (f16) W_p[c][h]   (p: 0=K 1=Q 2=V)
// Coalesced f16 writes (c fastest); W is tiny (0.75 MB) so strided reads are
// L2-trivial.
// ---------------------------------------------------------------------------
__global__ void wtrans_kernel(const float* __restrict__ Wk,
                              const float* __restrict__ Wq,
                              const float* __restrict__ Wv,
                              _Float16* __restrict__ Wt3) {
  const int idx = blockIdx.x * blockDim.x + threadIdx.x;
  if (idx >= 3 * HEADD * DMODEL) return;
  const int p = idx >> 16;          // 64*1024 per projection
  const int r = idx & 0xFFFF;
  const int h = r >> 10;
  const int c = r & 1023;
  const float* W = (p == 0) ? Wk : ((p == 1) ? Wq : Wv);
  Wt3[idx] = (_Float16)W[c * HEADD + h];
}

// ---------------------------------------------------------------------------
// Kernel 1: fused QKV projection. Block = 256 (8 waves), each wave owns a
// 16-row tile; grid = (CTX/128, NB). x is read exactly once. W chunks
// (3 projections x 64 x CCH f16 = 48 KB) are staged to LDS with
// global_load_async_to_lds_b128, double-buffered so staging of chunk i+1
// overlaps the 48 WMMAs/wave of chunk i.
// Dynamic LDS = 2 * 3*64*CCH*2 = 96 KB.
// ---------------------------------------------------------------------------
__global__ void qkv_proj_fused(const float* __restrict__ x,
                               const _Float16* __restrict__ Wt3,
                               _Float16* __restrict__ Kf,
                               _Float16* __restrict__ Qf,
                               _Float16* __restrict__ Vt) {
  extern __shared__ __align__(32) _Float16 Wbuf[];   // [2][3*HEADD*CCH]
  const int BUFH = 3 * HEADD * CCH;                  // halves per buffer

  const int b    = blockIdx.y;
  const int tid  = threadIdx.x;
  const int wave = tid >> 5;
  const int lane = tid & 31;
  const int hi   = lane >> 4;
  const int li   = lane & 15;
  const int m0   = (blockIdx.x * 8 + wave) * 16;

  v8f acc[3][4];
#pragma unroll
  for (int p = 0; p < 3; ++p)
#pragma unroll
    for (int n = 0; n < 4; ++n) acc[p][n] = vzero8();

  const float* xrow = x + ((size_t)b * CTX + (m0 + li)) * DMODEL;

  // ---- async stage of one W chunk into LDS buffer `buf` -------------------
  auto stage = [&](int buf, int c0) {
    const int nq = BUFH / 8;                 // 16B transfers (b128)
#pragma unroll 2
    for (int q = tid; q < nq; q += 256) {
      const int off = q << 3;                // halves
      const int p   = off >> 13;             // / (64*CCH)
      const int rem = off & 8191;
      const int h   = rem >> 7;              // / CCH
      const int c   = rem & (CCH - 1);
      const _Float16* g = Wt3 + (((size_t)p * HEADD + h) << 10) + c0 + c;
      const unsigned lds = (unsigned)(uintptr_t)(&Wbuf[buf * BUFH + off]);
      asm volatile("global_load_async_to_lds_b128 %0, %1, off"
                   :: "v"(lds), "v"(g) : "memory");
    }
  };

  stage(0, 0);
  asm volatile("s_wait_asynccnt 0x0" ::: "memory");
  __syncthreads();

  const int NCH = DMODEL / CCH;              // 8 chunks
  for (int ci = 0; ci < NCH; ++ci) {
    const int cur = ci & 1;
    if (ci + 1 < NCH) stage(1 - cur, (ci + 1) * CCH);   // overlap next stage

    const _Float16* wb = &Wbuf[cur * BUFH];
    for (int cc = 0; cc < CCH; cc += 32) {
      // A-fragment: lane<16 -> K{0..15}; lane>=16 -> K{16..31}
      const float* pA = xrow + ci * CCH + cc + hi * 16;
      f32x4 a0 = *(const f32x4*)(pA + 0);
      f32x4 a1 = *(const f32x4*)(pA + 4);
      f32x4 a2 = *(const f32x4*)(pA + 8);
      f32x4 a3 = *(const f32x4*)(pA + 12);
      v16h A;
#pragma unroll
      for (int j = 0; j < 4; ++j) {
        A[j]      = (_Float16)a0[j];
        A[4 + j]  = (_Float16)a1[j];
        A[8 + j]  = (_Float16)a2[j];
        A[12 + j] = (_Float16)a3[j];
      }
#pragma unroll
      for (int p = 0; p < 3; ++p) {
#pragma unroll
        for (int n = 0; n < 4; ++n) {
          const _Float16* bp =
              wb + ((p * HEADD + n * 16 + li) * CCH + cc + hi * 16);
          v16h B = *(const v16h*)bp;        // 2x ds_load_b128
          acc[p][n] = wmma_f16(A, B, acc[p][n]);
        }
      }
    }

    if (ci + 1 < NCH) {
      asm volatile("s_wait_asynccnt 0x0" ::: "memory");
      __syncthreads();                       // next buffer ready, all readers done
    }
  }

  // C layout: VGPR r, lane -> (M = m0 + r + 8*hi, N = n*16 + li)
#pragma unroll
  for (int p = 0; p < 3; ++p) {
    const float scale = (p == 1) ? 0.03125f : 1.0f;   // fold c^-0.5 into Q
#pragma unroll
    for (int n = 0; n < 4; ++n) {
      const int h = n * 16 + li;
#pragma unroll
      for (int r = 0; r < 8; ++r) {
        const int M = m0 + r + 8 * hi;
        const float v = acc[p][n][r] * scale;
        if (p == 2) {
          Vt[((size_t)b * HEADD + h) * CTX + M] = (_Float16)v;   // transposed
        } else {
          _Float16* dst = (p == 0) ? Kf : Qf;
          dst[((size_t)b * CTX + M) * HEADD + h] = (_Float16)v;  // row-major
        }
      }
    }
  }
}

// ---------------------------------------------------------------------------
// Kernel 2: flash attention. Each wave owns a 16-row query tile; kv processed
// 32 at a time (two 16x16 score tiles -> one 16x32 P A-fragment).
// grid = (CTX/128, NB); block = 256 (8 waves, no cross-wave sync).
// ---------------------------------------------------------------------------
__global__ void attn_kernel(const _Float16* __restrict__ Qf,
                            const _Float16* __restrict__ Kf,
                            const _Float16* __restrict__ Vt,
                            float* __restrict__ out) {
  __shared__ __align__(32) _Float16 pbuf[8][16 * 32];  // per-wave P staging

  const int tid  = threadIdx.x;
  const int wave = tid >> 5;
  const int lane = tid & 31;
  const int hi   = lane >> 4;
  const int li   = lane & 15;
  const int b    = blockIdx.y;
  const int m0   = (blockIdx.x * 8 + wave) * 16;

  const _Float16* Qb = Qf + (size_t)b * CTX * HEADD;
  const _Float16* Kb = Kf + (size_t)b * CTX * HEADD;
  const _Float16* Vb = Vt + (size_t)b * HEADD * CTX;

  // Q A-fragments for contraction chunks [0,32) and [32,64)
  v16h qa[2];
#pragma unroll
  for (int s = 0; s < 2; ++s) {
    const _Float16* p = Qb + (size_t)(m0 + li) * HEADD + s * 32 + hi * 16;
    v8h lo = *(const v8h*)(p);
    v8h hh = *(const v8h*)(p + 8);
    v16h A;
#pragma unroll
    for (int j = 0; j < 8; ++j) { A[j] = lo[j]; A[8 + j] = hh[j]; }
    qa[s] = A;
  }

  float mrow[8], lrow[8];
#pragma unroll
  for (int r = 0; r < 8; ++r) { mrow[r] = -INFINITY; lrow[r] = 0.0f; }
  v8f o[4];
#pragma unroll
  for (int n = 0; n < 4; ++n) o[n] = vzero8();

  _Float16* pb = pbuf[wave];

  for (int kv0 = 0; kv0 <= m0 + 15; kv0 += 32) {
    // ---- scores: two 16x16 tiles, contraction over 64 features ----
    v8f sfrag[2];
#pragma unroll
    for (int s = 0; s < 2; ++s) {
      const _Float16* kp =
          Kb + (size_t)(kv0 + 16 * s + li) * HEADD + hi * 16;
      v16h B0 = *(const v16h*)(kp);        // feats {0..15|16..31}
      v16h B1 = *(const v16h*)(kp + 32);   // feats {32..47|48..63}
      v8f z = vzero8();
      z = wmma_f16(qa[0], B0, z);
      z = wmma_f16(qa[1], B1, z);
      sfrag[s] = z;
    }

    // ---- causal mask (only near the diagonal tiles) ----
    if (kv0 + 31 > m0) {
#pragma unroll
      for (int s = 0; s < 2; ++s) {
        const int N = kv0 + 16 * s + li;
#pragma unroll
        for (int r = 0; r < 8; ++r) {
          const int M = m0 + r + 8 * hi;
          if (N > M) sfrag[s][r] = -INFINITY;
        }
      }
    }

    // ---- online softmax ----
#pragma unroll
    for (int r = 0; r < 8; ++r) {
      float t  = rowmax16(fmaxf(sfrag[0][r], sfrag[1][r]));
      float mn = fmaxf(mrow[r], t);
      float alpha = __expf(mrow[r] - mn);
      mrow[r] = mn;
      float p0 = __expf(sfrag[0][r] - mn);
      float p1 = __expf(sfrag[1][r] - mn);
      sfrag[0][r] = p0;
      sfrag[1][r] = p1;
      lrow[r] = lrow[r] * alpha + rowsum16(p0 + p1);
#pragma unroll
      for (int n = 0; n < 4; ++n) o[n][r] *= alpha;
    }

    // ---- C-layout -> A-layout via per-wave LDS tile (same-wave DS is
    //      in-order; no block barrier, waves have different trip counts) ----
#pragma unroll
    for (int r = 0; r < 8; ++r) {
      const int row = r + 8 * hi;
      pb[row * 32 + li]      = (_Float16)sfrag[0][r];
      pb[row * 32 + 16 + li] = (_Float16)sfrag[1][r];
    }
    asm volatile("" ::: "memory");
    const _Float16* pp = pb + li * 32 + hi * 16;
    v8h plo = *(const v8h*)(pp);
    v8h phi = *(const v8h*)(pp + 8);
    v16h PA;
#pragma unroll
    for (int j = 0; j < 8; ++j) { PA[j] = plo[j]; PA[8 + j] = phi[j]; }

    // ---- O += P @ V  (V stored transposed: contraction over t contiguous)
#pragma unroll
    for (int n = 0; n < 4; ++n) {
      const _Float16* vp =
          Vb + (size_t)(n * 16 + li) * CTX + kv0 + hi * 16;
      v16h B = *(const v16h*)vp;
      o[n] = wmma_f16(PA, B, o[n]);
    }
  }

  // ---- finalize: divide by row sums, store f32 ----
#pragma unroll
  for (int r = 0; r < 8; ++r) {
    const float inv = 1.0f / lrow[r];
    const int M = m0 + r + 8 * hi;
    float* orow = out + ((size_t)b * CTX + M) * HEADD;
#pragma unroll
    for (int n = 0; n < 4; ++n) orow[n * 16 + li] = o[n][r] * inv;
  }
}

// ---------------------------------------------------------------------------
extern "C" void kernel_launch(void* const* d_in, const int* in_sizes, int n_in,
                              void* d_out, int out_size, void* d_ws,
                              size_t ws_size, hipStream_t stream) {
  (void)in_sizes; (void)n_in; (void)out_size;

  const float* x  = (const float*)d_in[0];
  const float* Wk = (const float*)d_in[1];
  const float* Wq = (const float*)d_in[2];
  const float* Wv = (const float*)d_in[3];
  float* out = (float*)d_out;

  const size_t per = (size_t)NB * CTX * HEADD;        // f16 elems per tensor
  const size_t wtr = (size_t)3 * HEADD * DMODEL;      // transposed weights
  if (ws_size < (3 * per + wtr) * sizeof(_Float16)) return;
  _Float16* Kf  = (_Float16*)d_ws;
  _Float16* Qf  = Kf + per;
  _Float16* Vt  = Qf + per;
  _Float16* Wt3 = Vt + per;

  wtrans_kernel<<<(3 * HEADD * DMODEL + 255) / 256, 256, 0, stream>>>(
      Wk, Wq, Wv, Wt3);

  dim3 g1(CTX / 128, NB);
  const size_t smem1 = (size_t)2 * 3 * HEADD * CCH * sizeof(_Float16); // 96 KB
  qkv_proj_fused<<<g1, 256, smem1, stream>>>(x, Wt3, Kf, Qf, Vt);

  dim3 g2(CTX / 128, NB);
  attn_kernel<<<g2, 256, 0, stream>>>(Qf, Kf, Vt, out);
}
